// NAFS_26310969655366
// MI455X (gfx1250) — matrix operand order repaired and checked by reading, not compiled, not run on hardware
//
#include <hip/hip_runtime.h>
#include <math.h>

#define EPSF 1e-8f
#define LAMF 20.0f

constexpr int NI = 128;   // images
constexpr int NR = 48;    // regions per image
constexpr int NT = 64;    // texts
constexpr int NW = 100;   // words per text
constexpr int ND = 768;   // feature dim
constexpr int WP = 112;   // words padded to 7*16
constexpr int WK = 128;   // K-pad for stage2a (attn rows)
constexpr int RK = 64;    // K-pad for stage2b (attn2 cols)

typedef __attribute__((ext_vector_type(16))) __bf16 v16bf;
typedef __attribute__((ext_vector_type(8)))  float  v8f;

__device__ __forceinline__ unsigned short f2bf(float x){
  union { float f; unsigned u; } c; c.f = x;
  unsigned u = c.u;
  u = (u + 0x7FFFu + ((u >> 16) & 1u)) >> 16;  // RNE
  return (unsigned short)u;
}
__device__ __forceinline__ float bf2f(unsigned short hh){
  union { unsigned u; float f; } c; c.u = ((unsigned)hh) << 16;
  return c.f;
}

// A-operand frag (16x32 bf16). Per ISA layout: lane-half h holds K chunks
// [k0+8h .. +7] and [k0+16+8h .. +7]; row = m0 + (lane&15) (caller bakes row
// into `row` pointer).
__device__ __forceinline__ v16bf loadA(const unsigned short* row, int k0, int h){
  union { v16bf v; uint4 u[2]; } f;
  f.u[0] = *reinterpret_cast<const uint4*>(row + k0 + 8*h);
  f.u[1] = *reinterpret_cast<const uint4*>(row + k0 + 16 + 8*h);
  return f.v;
}
// B-operand frag (32x16 bf16): lane holds col n=lane&15, K chunks
// [k0+8h .. +7] and [k0+16+8h .. +7] (caller bakes the K-contiguous column pointer).
__device__ __forceinline__ v16bf loadB(const unsigned short* col, int k0, int h){
  union { v16bf v; uint4 u[2]; } f;
  f.u[0] = *reinterpret_cast<const uint4*>(col + k0 + 8*h);
  f.u[1] = *reinterpret_cast<const uint4*>(col + k0 + 16 + 8*h);
  return f.v;
}

#define WMMA_BF16(A,B,C) __builtin_amdgcn_wmma_f32_16x16x32_bf16(false,(A),false,(B),(short)0,(C),false,false)

// ---------------- precompute: normalize imgQ -> bf16, imgV norms + transposed bf16 copy
__global__ __launch_bounds__(256)
void prep_img(const float* __restrict__ imgQ, const float* __restrict__ imgV,
              unsigned short* __restrict__ Qn, unsigned short* __restrict__ ViT,
              float* __restrict__ nIV)
{
  const int bid = blockIdx.x;              // i*NR + r
  const int i = bid / NR, r = bid % NR;
  const float* q = imgQ + (size_t)bid * ND;
  const float* v = imgV + (size_t)bid * ND;
  float s1 = 0.f, s2 = 0.f;
  for (int d = threadIdx.x; d < ND; d += 256) {
    float a = q[d]; s1 += a*a;
    float b = v[d]; s2 += b*b;
  }
  for (int off = 16; off; off >>= 1) { s1 += __shfl_xor(s1, off, 32); s2 += __shfl_xor(s2, off, 32); }
  __shared__ float sb1[8], sb2[8];
  if ((threadIdx.x & 31) == 0) { sb1[threadIdx.x >> 5] = s1; sb2[threadIdx.x >> 5] = s2; }
  __syncthreads();
  float t1 = 0.f, t2 = 0.f;
  #pragma unroll
  for (int k = 0; k < 8; ++k) { t1 += sb1[k]; t2 += sb2[k]; }
  const float scale = 1.f / (sqrtf(t1) + EPSF);
  if (threadIdx.x == 0) nIV[bid] = sqrtf(t2);
  for (int d = threadIdx.x; d < ND; d += 256) {
    Qn[(size_t)bid * ND + d] = f2bf(q[d] * scale);
    ViT[((size_t)i * ND + d) * RK + r] = f2bf(v[d]);
  }
}

// ---------------- precompute: normalize textK -> bf16 (padded rows), textV norms + transposed copy
__global__ __launch_bounds__(256)
void prep_txt(const float* __restrict__ txtK, const float* __restrict__ txtV,
              unsigned short* __restrict__ Kn, unsigned short* __restrict__ VtT,
              float* __restrict__ nTV)
{
  const int bid = blockIdx.x;              // t*NW + w
  const int t = bid / NW, w = bid % NW;
  const float* k = txtK + (size_t)bid * ND;
  const float* v = txtV + (size_t)bid * ND;
  float s1 = 0.f, s2 = 0.f;
  for (int d = threadIdx.x; d < ND; d += 256) {
    float a = k[d]; s1 += a*a;
    float b = v[d]; s2 += b*b;
  }
  for (int off = 16; off; off >>= 1) { s1 += __shfl_xor(s1, off, 32); s2 += __shfl_xor(s2, off, 32); }
  __shared__ float sb1[8], sb2[8];
  if ((threadIdx.x & 31) == 0) { sb1[threadIdx.x >> 5] = s1; sb2[threadIdx.x >> 5] = s2; }
  __syncthreads();
  float t1 = 0.f, t2 = 0.f;
  #pragma unroll
  for (int kk = 0; kk < 8; ++kk) { t1 += sb1[kk]; t2 += sb2[kk]; }
  const float scale = 1.f / (sqrtf(t1) + EPSF);
  if (threadIdx.x == 0) nTV[(size_t)t * WP + w] = sqrtf(t2);
  for (int d = threadIdx.x; d < ND; d += 256) {
    Kn[((size_t)t * WP + w) * ND + d] = f2bf(k[d] * scale);
    VtT[((size_t)t * ND + d) * WK + w] = f2bf(v[d]);
  }
}

// ---------------- fused per-(t,i) kernel: 7 waves ----------------
__global__ __launch_bounds__(224)
void nafs_main(const unsigned short* __restrict__ Qn,
               const unsigned short* __restrict__ Kn,
               const unsigned short* __restrict__ VtT,
               const unsigned short* __restrict__ ViT,
               const float* __restrict__ nIV,
               const float* __restrict__ nTV,
               const float* __restrict__ imgV,
               const float* __restrict__ textV,
               const int* __restrict__ tlen,
               float* __restrict__ out)
{
  const int i    = blockIdx.x;
  const int t    = blockIdx.y;
  const int tid  = threadIdx.x;
  const int lane = tid & 31;
  const int wave = tid >> 5;        // 0..6
  const int h    = lane >> 4;       // lane half
  const int l15  = lane & 15;

  __shared__ __align__(16) float sA[WP * 52];               // leaky_relu(sim), 112x48 (stride 52)
  __shared__ __align__(16) unsigned short attn1s[NR * WK];  // branch-1 attn, 48x128 bf16
  __shared__ __align__(16) unsigned short attn2s[WP * RK];  // branch-2 attn, 112x64 bf16
  __shared__ float rnw[WP];     // per-w norm over r; reused for t2i partials
  __shared__ float wnr[NR];     // per-r norm over w; reused for i2t partials
  __shared__ float nsq1[NR], dot1[NR];
  __shared__ float nsq2[WP], dot2[WP];

  const int len = tlen[t];
  const float Lf = (float)len;

  // ===== Stage 1: sim = Kn[t] (112x768) x Qn[i]^T  via WMMA bf16 =====
  {
    const unsigned short* Kt = Kn + ((size_t)t * WP) * ND;
    const unsigned short* Qi = Qn + ((size_t)i * NR) * ND;
    const unsigned short* aRow = Kt + (size_t)(wave * 16 + l15) * ND;
    const unsigned short* b0 = Qi + (size_t)l15 * ND;
    const unsigned short* b1 = b0 + (size_t)16 * ND;
    const unsigned short* b2 = b0 + (size_t)32 * ND;
    v8f c0 = {0.f,0.f,0.f,0.f,0.f,0.f,0.f,0.f};
    v8f c1 = c0, c2 = c0;
    for (int k0 = 0; k0 < ND; k0 += 32) {
      v16bf a   = loadA(aRow, k0, h);
      v16bf bb0 = loadB(b0,  k0, h);
      v16bf bb1 = loadB(b1,  k0, h);
      v16bf bb2 = loadB(b2,  k0, h);
      c0 = WMMA_BF16(a, bb0, c0);
      c1 = WMMA_BF16(a, bb1, c1);
      c2 = WMMA_BF16(a, bb2, c2);
    }
    #pragma unroll
    for (int v = 0; v < 8; ++v) {
      const int w = wave * 16 + 8 * h + v;
      float x0 = c0[v]; x0 = x0 > 0.f ? x0 : 0.1f * x0;
      float x1 = c1[v]; x1 = x1 > 0.f ? x1 : 0.1f * x1;
      float x2 = c2[v]; x2 = x2 > 0.f ? x2 : 0.1f * x2;
      sA[w * 52 + l15]      = x0;
      sA[w * 52 + 16 + l15] = x1;
      sA[w * 52 + 32 + l15] = x2;
    }
  }
  __syncthreads();

  // ===== norms for the two branches =====
  if (tid < WP) {
    const int w = tid; float s = 0.f;
    for (int r = 0; r < NR; ++r) { float v = sA[w * 52 + r]; s += v * v; }
    rnw[w] = sqrtf(s);
  } else if (tid < WP + NR) {
    const int r = tid - WP; float s = 0.f;
    for (int w = 0; w < len; ++w) { float v = sA[w * 52 + r]; s += v * v; }
    wnr[r] = sqrtf(s);
  }
  __syncthreads();

  // ===== softmaxes + Fermi filtering =====
  if (tid < NR) {                               // branch 1, per region r
    const int r = tid;
    float mx = -1e30f;
    for (int w = 0; w < len; ++w) {
      float x = sA[w * 52 + r] * LAMF / (rnw[w] + EPSF);
      mx = fmaxf(mx, x);
    }
    float se = 0.f;
    for (int w = 0; w < len; ++w) {
      float x = sA[w * 52 + r] * LAMF / (rnw[w] + EPSF);
      se += expf(x - mx);
    }
    const float inv = 1.f / se;
    float ts = 0.f;
    for (int w = 0; w < len; ++w) {
      float x = sA[w * 52 + r] * LAMF / (rnw[w] + EPSF);
      float av = expf(x - mx) * inv;
      float kv = (av * Lf > 1.f) ? av : 0.f;    // funcF > 0
      ts += kv;
      attn1s[r * WK + w] = f2bf(kv);
    }
    for (int w = len; w < WK; ++w) attn1s[r * WK + w] = 0;
    const float sc = ts > 0.f ? 1.f / ts : 1.f; // _safe_div
    for (int w = 0; w < len; ++w)
      attn1s[r * WK + w] = f2bf(bf2f(attn1s[r * WK + w]) * sc);
  } else if (tid < WP) {                        // idle lanes zero accumulators
    const int idx = tid - NR;                   // 0..63
    if (idx < NR) { nsq1[idx] = 0.f; dot1[idx] = 0.f; }
    const int a = 2 * idx, b = 2 * idx + 1;
    if (a < WP) { nsq2[a] = 0.f; dot2[a] = 0.f; }
    if (b < WP) { nsq2[b] = 0.f; dot2[b] = 0.f; }
  } else {                                      // branch 2, per word w
    const int w = tid - WP;                     // 0..111
    if (w >= len) {
      for (int r = 0; r < RK; ++r) attn2s[w * RK + r] = 0;
    } else {
      float mx = -1e30f;
      for (int r = 0; r < NR; ++r) {
        float x = sA[w * 52 + r] * LAMF / (wnr[r] + EPSF);
        mx = fmaxf(mx, x);
      }
      float se = 0.f;
      for (int r = 0; r < NR; ++r) {
        float x = sA[w * 52 + r] * LAMF / (wnr[r] + EPSF);
        se += expf(x - mx);
      }
      const float inv = 1.f / se;
      float ts = 0.f;
      for (int r = 0; r < NR; ++r) {
        float x = sA[w * 52 + r] * LAMF / (wnr[r] + EPSF);
        float av = expf(x - mx) * inv;
        float kv = (av * 48.f > 1.f) ? av : 0.f;
        ts += kv;
        attn2s[w * RK + r] = f2bf(kv);
      }
      for (int r = NR; r < RK; ++r) attn2s[w * RK + r] = 0;
      const float sc = ts > 0.f ? 1.f / ts : 1.f;
      for (int r = 0; r < NR; ++r)
        attn2s[w * RK + r] = f2bf(bf2f(attn2s[w * RK + r]) * sc);
    }
  }
  __syncthreads();

  // ===== Stage 2a: weiText = attn1(48x128) @ textV(128x768); fold into nsq1/dot1 =====
  for (int mt = 0; mt < 3; ++mt) {
    float accN[8], accD[8];
    #pragma unroll
    for (int v = 0; v < 8; ++v) { accN[v] = 0.f; accD[v] = 0.f; }
    const unsigned short* aRow = attn1s + (size_t)(mt * 16 + l15) * WK;
    const int rbase = mt * 16 + 8 * h;
    for (int nt = wave; nt < 48; nt += 7) {
      const int d = nt * 16 + l15;
      const unsigned short* bCol = VtT + ((size_t)t * ND + d) * WK;
      v8f c = {0.f,0.f,0.f,0.f,0.f,0.f,0.f,0.f};
      #pragma unroll
      for (int ks = 0; ks < 4; ++ks) {
        v16bf a = loadA(aRow, ks * 32, h);
        v16bf b = loadB(bCol, ks * 32, h);
        c = WMMA_BF16(a, b, c);
      }
      const float* iv = imgV + ((size_t)i * NR + rbase) * ND + d;
      #pragma unroll
      for (int v = 0; v < 8; ++v) {
        const float val = c[v];
        accN[v] += val * val;
        accD[v] += val * iv[(size_t)v * ND];
      }
    }
    #pragma unroll
    for (int v = 0; v < 8; ++v) {
      float n = accN[v], dd = accD[v];
      for (int off = 8; off; off >>= 1) {
        n  += __shfl_xor(n,  off, 32);
        dd += __shfl_xor(dd, off, 32);
      }
      if (l15 == 0) {
        const int r = rbase + v;
        atomicAdd(&nsq1[r], n);
        atomicAdd(&dot1[r], dd);
      }
    }
  }

  // ===== Stage 2b: weiImage = attn2(112x64) @ imgV(64x768); fold into nsq2/dot2 =====
  for (int mt = 0; mt < 7; ++mt) {
    float accN[8], accD[8];
    #pragma unroll
    for (int v = 0; v < 8; ++v) { accN[v] = 0.f; accD[v] = 0.f; }
    const unsigned short* aRow = attn2s + (size_t)(mt * 16 + l15) * RK;
    const int wbase = mt * 16 + 8 * h;
    for (int nt = wave; nt < 48; nt += 7) {
      const int d = nt * 16 + l15;
      const unsigned short* bCol = ViT + ((size_t)i * ND + d) * RK;
      v8f c = {0.f,0.f,0.f,0.f,0.f,0.f,0.f,0.f};
      #pragma unroll
      for (int ks = 0; ks < 2; ++ks) {
        v16bf a = loadA(aRow, ks * 32, h);
        v16bf b = loadB(bCol, ks * 32, h);
        c = WMMA_BF16(a, b, c);
      }
      #pragma unroll
      for (int v = 0; v < 8; ++v) {
        const float val = c[v];
        accN[v] += val * val;
        const int w = wbase + v;
        const float tv = (w < NW) ? textV[((size_t)t * NW + w) * ND + d] : 0.f;
        accD[v] += val * tv;
      }
    }
    #pragma unroll
    for (int v = 0; v < 8; ++v) {
      float n = accN[v], dd = accD[v];
      for (int off = 8; off; off >>= 1) {
        n  += __shfl_xor(n,  off, 32);
        dd += __shfl_xor(dd, off, 32);
      }
      if (l15 == 0) {
        const int w = wbase + v;
        atomicAdd(&nsq2[w], n);
        atomicAdd(&dot2[w], dd);
      }
    }
  }
  __syncthreads();

  // ===== cosine similarity + final reductions =====
  if (tid < NR) {
    const int r = tid;
    const float nw = sqrtf(nsq1[r]);
    const float innw = 1.f / (nw + EPSF);
    const float numv = dot1[r] * innw;
    const float den = fmaxf(nIV[(size_t)i * NR + r] * nw * innw, EPSF);
    wnr[r] = numv / den;
  } else if (tid >= WP && tid < 2 * WP) {
    const int w = tid - WP;
    float cv = 0.f;
    if (w < len) {
      const float nw = sqrtf(nsq2[w]);
      const float innw = 1.f / (nw + EPSF);
      const float numv = dot2[w] * innw;
      const float den = fmaxf(nTV[(size_t)t * WP + w] * nw * innw, EPSF);
      cv = numv / den;
    }
    rnw[w] = cv;
  }
  __syncthreads();
  if (tid == 0) {
    float s = 0.f;
    for (int r = 0; r < NR; ++r) s += wnr[r];
    out[(size_t)i * NT + t] = s / 48.f;                       // i2t.T
  }
  if (tid == 32) {
    float s = 0.f;
    for (int w = 0; w < WP; ++w) s += rnw[w];
    out[(size_t)NI * NT + (size_t)i * NT + t] = s / Lf;       // t2i.T
  }
}

extern "C" void kernel_launch(void* const* d_in, const int* in_sizes, int n_in,
                              void* d_out, int out_size, void* d_ws, size_t ws_size,
                              hipStream_t stream)
{
  (void)in_sizes; (void)n_in; (void)out_size; (void)ws_size;
  const float* imgQ = (const float*)d_in[0];
  const float* imgV = (const float*)d_in[1];
  const float* txtK = (const float*)d_in[2];
  const float* txtV = (const float*)d_in[3];
  const int*   tlen = (const int*)d_in[4];
  float* out = (float*)d_out;

  char* ws = (char*)d_ws;
  size_t off = 0;
  auto carve = [&](size_t bytes) -> void* {
    void* p = ws + off;
    off = (off + bytes + 255) & ~(size_t)255;
    return p;
  };
  unsigned short* Qn  = (unsigned short*)carve((size_t)NI * NR * ND * 2); // normalized imgQ bf16
  unsigned short* Kn  = (unsigned short*)carve((size_t)NT * WP * ND * 2); // normalized txtK bf16 (row-padded)
  unsigned short* VtT = (unsigned short*)carve((size_t)NT * ND * WK * 2); // txtV transposed bf16 (K-padded)
  unsigned short* ViT = (unsigned short*)carve((size_t)NI * ND * RK * 2); // imgV transposed bf16 (K-padded)
  float* nIV = (float*)carve((size_t)NI * NR * 4);
  float* nTV = (float*)carve((size_t)NT * WP * 4);

  // zero the padded bf16 buffers (pad rows/cols must be 0 for WMMA K-padding)
  hipMemsetAsync(Kn,  0, (size_t)NT * WP * ND * 2, stream);
  hipMemsetAsync(VtT, 0, (size_t)NT * ND * WK * 2, stream);
  hipMemsetAsync(ViT, 0, (size_t)NI * ND * RK * 2, stream);

  prep_img<<<NI * NR, 256, 0, stream>>>(imgQ, imgV, Qn, ViT, nIV);
  prep_txt<<<NT * NW, 256, 0, stream>>>(txtK, txtV, Kn, VtT, nTV);
  nafs_main<<<dim3(NI, NT), 224, 0, stream>>>(Qn, Kn, VtT, ViT, nIV, nTV,
                                              imgV, txtV, tlen, out);
}
